// MultiHeadSelfAttention_12352325943563
// MI455X (gfx1250) — matrix operand order repaired and checked by reading, not compiled, not run on hardware
//
#include <hip/hip_runtime.h>
#include <hip/hip_bf16.h>

#define EMBED 1024
#define SEQ   2048
#define BATCH 2
#define HEADS 16
#define HD    64
#define ROWS  (BATCH*SEQ)   /* 4096 */
#define QKVC  (3*EMBED)     /* 3072 */
#define ATTN_SCALE 0.125f   /* 64^-0.5 */

typedef __attribute__((ext_vector_type(16))) __bf16 v16bf;
typedef __attribute__((ext_vector_type(8)))  float  v8f;
typedef __attribute__((ext_vector_type(4)))  unsigned int v4u;
typedef __attribute__((ext_vector_type(8)))  int v8i;
typedef __attribute__((ext_vector_type(4)))  int v4i;

struct __align__(16) U4 { unsigned int x, y, z, w; };

union Frag16 {
    v16bf v;
    U4 q[2];
    unsigned short us[16];
};

__device__ __forceinline__ unsigned short f2bf(float f) {
    unsigned int u = __float_as_uint(f);
    u += 0x7FFFu + ((u >> 16) & 1u);      // round to nearest even
    return (unsigned short)(u >> 16);
}

__device__ __forceinline__ v8f v8f_zero() {
    v8f r = {0.f,0.f,0.f,0.f,0.f,0.f,0.f,0.f};
    return r;
}

// ---- Tensor Data Mover: 2D bf16 tile [64 rows x 64 elems] global -> LDS.
// Row stride in memory = HD elements; LDS gets 16B padding after every 128B
// (pad_interval code 4 = 32 DWORDs, pad_amount code 3 = 4 DWORDs) so the
// LDS row stride is 144B = SKP(=72) elements, matching the compute layout.
__device__ __forceinline__ void tdm_load_k_tile(const unsigned short* gsrc,
                                                unsigned int lds_off) {
    unsigned long long ga = (unsigned long long)(size_t)gsrc;
    v4u g0;
    g0[0] = 1u;                                   // count=1, user mode
    g0[1] = lds_off;                              // lds_addr (bytes)
    g0[2] = (unsigned int)ga;                     // global_addr[31:0]
    g0[3] = (unsigned int)((ga >> 32) & 0x01FFFFFFull) | (2u << 30); // [56:32] | type=2
    v8i g1;
    g1[0] = (int)((1u << 16) | (1u << 20) | (4u << 22) | (3u << 25));
            // data_size=2B, pad_enable, pad_interval=32DW, pad_amount=4DW
    g1[1] = (int)((HD & 0xFFFFu) << 16);          // tensor_dim0[15:0]
    g1[2] = (int)(((unsigned)HD >> 16) | (((unsigned)SEQ & 0xFFFFu) << 16));
    g1[3] = (int)(((unsigned)SEQ >> 16) | (64u << 16));   // tile_dim0=64
    g1[4] = (int)(64u);                           // tile_dim1=64, tile_dim2=0
    g1[5] = (int)((unsigned)HD);                  // tensor_dim0_stride[31:0]
    g1[6] = 0;
    g1[7] = 0;
    v4i gz = {0, 0, 0, 0};
#if defined(__clang_major__) && (__clang_major__ >= 23)
    v8i gz8 = {0, 0, 0, 0, 0, 0, 0, 0};
    __builtin_amdgcn_tensor_load_to_lds(g0, g1, gz, gz, gz8, 0);
#else
    __builtin_amdgcn_tensor_load_to_lds(g0, g1, gz, gz, 0);
#endif
}

// ---------------------------------------------------------------- fp32->bf16
__global__ __launch_bounds__(256) void cvt_bf16_kernel(
        const float* __restrict__ src, unsigned short* __restrict__ dst, int n) {
    int i = blockIdx.x * blockDim.x + threadIdx.x;
    int stride = gridDim.x * blockDim.x;
    for (; i < n; i += stride) dst[i] = f2bf(src[i]);
}

// ---------------------------------------------------------------- QKV GEMM
// C[4096,3072] = Xbf[4096,1024] * Wbf^T + bias, scattered into qkvT
// [3][B][H][N][HD] bf16.  Block = 128 thr (4 waves); wave tile = 32x64
// (2 A-frags x 4 B-frags = 8 WMMA per k-step, B frags reused across rows).
__global__ __launch_bounds__(128) void qkv_gemm_kernel(
        const unsigned short* __restrict__ X,
        const unsigned short* __restrict__ W,
        const float* __restrict__ bias,
        unsigned short* __restrict__ qkvT) {
    const int lane = threadIdx.x & 31;
    const int wave = threadIdx.x >> 5;
    const int hi = lane >> 4;
    const int lo = lane & 15;
    const int row0 = blockIdx.y * 128 + wave * 32;
    const int col0 = blockIdx.x * 64;

    v8f acc[2][4];
#pragma unroll
    for (int r = 0; r < 2; ++r)
#pragma unroll
        for (int t = 0; t < 4; ++t) acc[r][t] = v8f_zero();

    const unsigned short* xrow0 = X + (size_t)(row0 + lo) * EMBED;
    const unsigned short* xrow1 = X + (size_t)(row0 + 16 + lo) * EMBED;
    for (int kb = 0; kb < EMBED; kb += 32) {
        Frag16 a0, a1;
        a0.q[0] = *(const U4*)(xrow0 + kb + hi * 8);
        a0.q[1] = *(const U4*)(xrow0 + kb + 16 + hi * 8);
        a1.q[0] = *(const U4*)(xrow1 + kb + hi * 8);
        a1.q[1] = *(const U4*)(xrow1 + kb + 16 + hi * 8);
#pragma unroll
        for (int t = 0; t < 4; ++t) {
            const unsigned short* wrow =
                W + (size_t)(col0 + t * 16 + lo) * EMBED + kb + hi * 8;
            Frag16 b;
            b.q[0] = *(const U4*)(wrow);
            b.q[1] = *(const U4*)(wrow + 16);
            acc[0][t] = __builtin_amdgcn_wmma_f32_16x16x32_bf16(
                false, a0.v, false, b.v, (short)0, acc[0][t], false, false);
            acc[1][t] = __builtin_amdgcn_wmma_f32_16x16x32_bf16(
                false, a1.v, false, b.v, (short)0, acc[1][t], false, false);
        }
    }

#pragma unroll
    for (int t = 0; t < 4; ++t) {
        const int col = col0 + t * 16 + lo;
        const float bv = bias[col];
        const int which = col >> 10;      // 0=q 1=k 2=v
        const int d = col & 1023;
        const int h = d >> 6;
        const int dd = d & 63;
#pragma unroll
        for (int r = 0; r < 2; ++r)
#pragma unroll
            for (int j = 0; j < 8; ++j) {
                const int row = row0 + r * 16 + j + 8 * hi;
                const int b = row >> 11;
                const int npos = row & 2047;
                const size_t off =
                    ((((size_t)which * BATCH + b) * HEADS + h) * SEQ + npos) * HD + dd;
                qkvT[off] = f2bf(acc[r][t][j] + bv);
            }
    }
}

// ---------------------------------------------------------------- attention
// Flash-style online-softmax attention per (b,h).  Block = 256 thr (8 waves),
// each wave owns 16 queries (128/block).  Key blocks of 64; K tile staged by
// the Tensor Data Mover, V tile staged transposed by the block.
#define SKP 72   /* padded LDS row stride in elements (16B mult, bank-friendly) */

__global__ __launch_bounds__(256) void attn_kernel(
        const unsigned short* __restrict__ qkvT,
        unsigned short* __restrict__ attnOut /* [B,N,EMBED] bf16 */) {
    const int bh = blockIdx.x;          // 0..31
    const int b  = bh >> 4;
    const int h  = bh & 15;
    const int qt = blockIdx.y;          // 0..15
    const int tid  = threadIdx.x;
    const int lane = tid & 31;
    const int wave = tid >> 5;
    const int hi = lane >> 4;
    const int lo = lane & 15;

    const size_t headOff  = (((size_t)b * HEADS + h) * SEQ) * HD;
    const size_t planeSz  = (size_t)BATCH * HEADS * SEQ * HD;
    const unsigned short* Q = qkvT + headOff;
    const unsigned short* K = qkvT + planeSz + headOff;
    const unsigned short* V = qkvT + 2 * planeSz + headOff;

    __shared__ unsigned short sK[64 * SKP];        // [key][d], filled by TDM
    __shared__ unsigned short sVt[64 * SKP];       // [d][key]  (transposed)
    __shared__ unsigned short sP[8][16 * SKP];     // per-wave P tile [m][key]

    const unsigned int sK_lds = (unsigned int)(size_t)(&sK[0]);
    const int q0 = qt * 128 + wave * 16;

    // Q fragments (held in registers for the whole kernel)
    Frag16 aq[2];
    {
        const unsigned short* qrow = Q + (size_t)(q0 + lo) * HD;
#pragma unroll
        for (int f = 0; f < 2; ++f) {
            aq[f].q[0] = *(const U4*)(qrow + f * 32 + hi * 8);
            aq[f].q[1] = *(const U4*)(qrow + f * 32 + 16 + hi * 8);
        }
    }

    float mrun[8], lrun[8];
    v8f o[4];
#pragma unroll
    for (int j = 0; j < 8; ++j) { mrun[j] = -3.0e38f; lrun[j] = 0.f; }
#pragma unroll
    for (int t = 0; t < 4; ++t) o[t] = v8f_zero();

    const int skey = tid >> 2;          // 0..63
    const int sd   = (tid & 3) * 16;    // 0,16,32,48

    for (int kb = 0; kb < SEQ; kb += 64) {
        __syncthreads();   // previous iteration done reading sK/sVt
        // ---- K tile: one wave issues a TDM descriptor (pads rows to SKP)
        if (wave == 0)
            tdm_load_k_tile(K + (size_t)kb * HD, sK_lds);
        // ---- V tile: all threads stage transposed
        {
            const unsigned short* vs = V + (size_t)(kb + skey) * HD + sd;
            Frag16 vv;
            vv.q[0] = *(const U4*)(vs);
            vv.q[1] = *(const U4*)(vs + 8);
#pragma unroll
            for (int i = 0; i < 16; ++i) sVt[(sd + i) * SKP + skey] = vv.us[i];

            if (kb + 64 < SEQ)   // prefetch next V tile (global_prefetch_b8)
                __builtin_prefetch(V + (size_t)(kb + 64 + skey) * HD + sd, 0, 0);
        }
        if (wave == 0)
            __builtin_amdgcn_s_wait_tensorcnt(0);
        __syncthreads();

        // ---- S = scale * Q K^T   (4 col tiles of 16 keys)
        v8f s[4];
#pragma unroll
        for (int t = 0; t < 4; ++t) {
            s[t] = v8f_zero();
#pragma unroll
            for (int f = 0; f < 2; ++f) {
                const unsigned short* kr = &sK[(t * 16 + lo) * SKP + f * 32 + hi * 8];
                Frag16 bk;
                bk.q[0] = *(const U4*)(kr);
                bk.q[1] = *(const U4*)(kr + 16);
                s[t] = __builtin_amdgcn_wmma_f32_16x16x32_bf16(
                    false, aq[f].v, false, bk.v, (short)0, s[t], false, false);
            }
#pragma unroll
            for (int j = 0; j < 8; ++j) s[t][j] *= ATTN_SCALE;
        }

        // ---- online softmax update (rows live across 16-lane groups)
        float sc[8];
#pragma unroll
        for (int j = 0; j < 8; ++j) {
            float tmax = fmaxf(fmaxf(s[0][j], s[1][j]), fmaxf(s[2][j], s[3][j]));
#pragma unroll
            for (int m = 1; m < 16; m <<= 1)
                tmax = fmaxf(tmax, __shfl_xor(tmax, m, 32));
            const float mnew = fmaxf(mrun[j], tmax);
            sc[j] = __expf(mrun[j] - mnew);
            float rsum = 0.f;
#pragma unroll
            for (int t = 0; t < 4; ++t) {
                s[t][j] = __expf(s[t][j] - mnew);
                rsum += s[t][j];
            }
#pragma unroll
            for (int m = 1; m < 16; m <<= 1)
                rsum += __shfl_xor(rsum, m, 32);
            lrun[j] = lrun[j] * sc[j] + rsum;
            mrun[j] = mnew;
        }
#pragma unroll
        for (int t = 0; t < 4; ++t)
#pragma unroll
            for (int j = 0; j < 8; ++j) o[t][j] *= sc[j];

        // ---- P: C-layout -> A-layout via wave-private LDS round trip
        unsigned short* pw = sP[wave];
#pragma unroll
        for (int t = 0; t < 4; ++t)
#pragma unroll
            for (int j = 0; j < 8; ++j)
                pw[(j + 8 * hi) * SKP + t * 16 + lo] = f2bf(s[t][j]);

        Frag16 ap[2];
        {
            const unsigned short* prow = pw + lo * SKP;
#pragma unroll
            for (int f = 0; f < 2; ++f) {
                ap[f].q[0] = *(const U4*)(prow + f * 32 + hi * 8);
                ap[f].q[1] = *(const U4*)(prow + f * 32 + 16 + hi * 8);
            }
        }

        // ---- O += P V   (4 hd-column tiles)
#pragma unroll
        for (int t = 0; t < 4; ++t) {
#pragma unroll
            for (int f = 0; f < 2; ++f) {
                const unsigned short* vr = &sVt[(t * 16 + lo) * SKP + f * 32 + hi * 8];
                Frag16 bv;
                bv.q[0] = *(const U4*)(vr);
                bv.q[1] = *(const U4*)(vr + 16);
                o[t] = __builtin_amdgcn_wmma_f32_16x16x32_bf16(
                    false, ap[f].v, false, bv.v, (short)0, o[t], false, false);
            }
        }
    }

    // ---- normalize and store to [B,N,EMBED] bf16 (head-interleaved cols)
    float inv[8];
#pragma unroll
    for (int j = 0; j < 8; ++j) inv[j] = 1.f / lrun[j];
#pragma unroll
    for (int t = 0; t < 4; ++t)
#pragma unroll
        for (int j = 0; j < 8; ++j) {
            const int row = q0 + j + 8 * hi;
            const int col = h * HD + t * 16 + lo;
            attnOut[((size_t)b * SEQ + row) * EMBED + col] = f2bf(o[t][j] * inv[j]);
        }
}

// ---------------------------------------------------------------- out proj
// out[4096,1024] (fp32) = A_bf[4096,1024] * Wproj^T + bias; wave tile 32x64.
__global__ __launch_bounds__(128) void proj_gemm_kernel(
        const unsigned short* __restrict__ A,
        const unsigned short* __restrict__ W,
        const float* __restrict__ bias,
        float* __restrict__ out) {
    const int lane = threadIdx.x & 31;
    const int wave = threadIdx.x >> 5;
    const int hi = lane >> 4;
    const int lo = lane & 15;
    const int row0 = blockIdx.y * 128 + wave * 32;
    const int col0 = blockIdx.x * 64;

    v8f acc[2][4];
#pragma unroll
    for (int r = 0; r < 2; ++r)
#pragma unroll
        for (int t = 0; t < 4; ++t) acc[r][t] = v8f_zero();

    const unsigned short* arow0 = A + (size_t)(row0 + lo) * EMBED;
    const unsigned short* arow1 = A + (size_t)(row0 + 16 + lo) * EMBED;
    for (int kb = 0; kb < EMBED; kb += 32) {
        Frag16 a0, a1;
        a0.q[0] = *(const U4*)(arow0 + kb + hi * 8);
        a0.q[1] = *(const U4*)(arow0 + kb + 16 + hi * 8);
        a1.q[0] = *(const U4*)(arow1 + kb + hi * 8);
        a1.q[1] = *(const U4*)(arow1 + kb + 16 + hi * 8);
#pragma unroll
        for (int t = 0; t < 4; ++t) {
            const unsigned short* wrow =
                W + (size_t)(col0 + t * 16 + lo) * EMBED + kb + hi * 8;
            Frag16 b;
            b.q[0] = *(const U4*)(wrow);
            b.q[1] = *(const U4*)(wrow + 16);
            acc[0][t] = __builtin_amdgcn_wmma_f32_16x16x32_bf16(
                false, a0.v, false, b.v, (short)0, acc[0][t], false, false);
            acc[1][t] = __builtin_amdgcn_wmma_f32_16x16x32_bf16(
                false, a1.v, false, b.v, (short)0, acc[1][t], false, false);
        }
    }

#pragma unroll
    for (int t = 0; t < 4; ++t) {
        const int col = col0 + t * 16 + lo;
        const float bv = bias[col];
#pragma unroll
        for (int r = 0; r < 2; ++r)
#pragma unroll
            for (int j = 0; j < 8; ++j) {
                const int row = row0 + r * 16 + j + 8 * hi;
                out[(size_t)row * EMBED + col] = acc[r][t][j] + bv;
            }
    }
}

// ---------------------------------------------------------------- launcher
extern "C" void kernel_launch(void* const* d_in, const int* in_sizes, int n_in,
                              void* d_out, int out_size, void* d_ws, size_t ws_size,
                              hipStream_t stream) {
    const float* x      = (const float*)d_in[0];
    const float* w_qkv  = (const float*)d_in[1];
    const float* b_qkv  = (const float*)d_in[2];
    const float* w_proj = (const float*)d_in[3];
    const float* b_proj = (const float*)d_in[4];
    float* out = (float*)d_out;

    char* ws = (char*)d_ws;
    // workspace layout (bytes)
    unsigned short* xb     = (unsigned short*)(ws);             //  8.0 MB @ 0
    unsigned short* wqkvb  = (unsigned short*)(ws +  8388608);  //  6.0 MB
    unsigned short* wprojb = (unsigned short*)(ws + 14680064);  //  2.0 MB
    unsigned short* qkvT   = (unsigned short*)(ws + 16777216);  // 24.0 MB
    unsigned short* attnb  = (unsigned short*)(ws + 41943040);  //  8.0 MB

    // 1) fp32 -> bf16 casts
    cvt_bf16_kernel<<<dim3(512), dim3(256), 0, stream>>>(x,      xb,     ROWS * EMBED);
    cvt_bf16_kernel<<<dim3(512), dim3(256), 0, stream>>>(w_qkv,  wqkvb,  QKVC * EMBED);
    cvt_bf16_kernel<<<dim3(256), dim3(256), 0, stream>>>(w_proj, wprojb, EMBED * EMBED);

    // 2) QKV projection -> head-major bf16 Q/K/V
    qkv_gemm_kernel<<<dim3(QKVC / 64, ROWS / 128), dim3(128), 0, stream>>>(
        xb, wqkvb, b_qkv, qkvT);

    // 3) flash attention (TDM-staged K tiles)
    attn_kernel<<<dim3(BATCH * HEADS, SEQ / 128), dim3(256), 0, stream>>>(
        qkvT, attnb);

    // 4) output projection (fp32 out)
    proj_gemm_kernel<<<dim3(EMBED / 64, ROWS / 128), dim3(128), 0, stream>>>(
        attnb, wprojb, b_proj, out);
}
